// IncrementalAttention_7009386627429
// MI455X (gfx1250) — matrix operand order repaired and checked by reading, not compiled, not run on hardware
//
#include <hip/hip_runtime.h>
#include <hip/hip_bf16.h>
#include <stddef.h>

// ---------------------------------------------------------------------------
// CDNA5 (gfx1250) wave32 WMMA implementation of a Llama attention block.
//   B=2, S=1024, H=4096, NH=32, NKV=8, HD=128, rope theta = 1e6
// Outputs (flat concat, bf16): out[B,S,H] ++ cache_k[B,S,NKV,HD] ++ cache_v[...]
// ---------------------------------------------------------------------------

typedef __attribute__((ext_vector_type(16))) __bf16 v16bf;
typedef __attribute__((ext_vector_type(8)))  __bf16 v8bf;
typedef __attribute__((ext_vector_type(4)))  __bf16 v4bf;
typedef __attribute__((ext_vector_type(8)))  float  v8f;
typedef unsigned short u16;
typedef unsigned int   u32;

union F16Frag { v16bf v; v8bf h[2]; };
union U4 { v4bf v; u16 s[4]; };

// Async global->LDS copy path (gfx1250 GLOBAL_LOAD_ASYNC_TO_LDS_B128, ASYNCcnt)
// Probe-confirmed: builtin exists; param0 is int-vec4 pointer in AS1 (global),
// param1 the LDS-side pointer, then two int immediates (offset, cpol).
#if defined(__AMDGCN__) && __has_builtin(__builtin_amdgcn_global_load_async_to_lds_b128) && __has_builtin(__builtin_amdgcn_s_wait_asynccnt)
#define USE_ASYNC_LDS 1
typedef int v4i_vs __attribute__((vector_size(16)));
typedef v4i_vs __attribute__((address_space(1))) as1_v4i;
typedef v4i_vs __attribute__((address_space(3))) as3_v4i;
#else
#define USE_ASYNC_LDS 0
#endif

static __device__ inline v8f vzero8() {
  v8f z;
#pragma unroll
  for (int i = 0; i < 8; ++i) z[i] = 0.0f;
  return z;
}

static __device__ inline v8f wmma_bf16(v16bf a, v16bf b, v8f c) {
  // v_wmma_f32_16x16x32_bf16 (neg_a, A, neg_b, B, c_mod, C, reuse_a, reuse_b)
  return __builtin_amdgcn_wmma_f32_16x16x32_bf16(false, a, false, b, (short)0, c,
                                                 false, false);
}

// ---- staging helpers: 16 elements from a global row -> LDS (bf16) ---------
static __device__ inline void stage16(const float* __restrict__ src,
                                      __bf16* __restrict__ dst) {
  const float4* s = (const float4*)src;
  float4 f0 = s[0], f1 = s[1], f2 = s[2], f3 = s[3];
  dst[0]  = (__bf16)f0.x; dst[1]  = (__bf16)f0.y; dst[2]  = (__bf16)f0.z; dst[3]  = (__bf16)f0.w;
  dst[4]  = (__bf16)f1.x; dst[5]  = (__bf16)f1.y; dst[6]  = (__bf16)f1.z; dst[7]  = (__bf16)f1.w;
  dst[8]  = (__bf16)f2.x; dst[9]  = (__bf16)f2.y; dst[10] = (__bf16)f2.z; dst[11] = (__bf16)f2.w;
  dst[12] = (__bf16)f3.x; dst[13] = (__bf16)f3.y; dst[14] = (__bf16)f3.z; dst[15] = (__bf16)f3.w;
}
static __device__ inline void stage16(const __bf16* __restrict__ src,
                                      __bf16* __restrict__ dst) {
  *(v8bf*)dst       = *(const v8bf*)src;
  *(v8bf*)(dst + 8) = *(const v8bf*)(src + 8);
}

// ---------------------------------------------------------------------------
// Tiled GEMM: C[M,N](bf16) = A[M,K] x W[K,N](bf16) (+ bias[N])
// BM=BN=128, BK=32; 256 threads = 8 waves; wave computes 32x64.
// LDS double-buffered: next tile stages while current tile's WMMAs run.
// W tile stored transposed [col][k] via packed ds_store_b64 (k-contiguous).
// ---------------------------------------------------------------------------
template <typename AT, bool HAS_BIAS>
__global__ __launch_bounds__(256) void gemm_wmma(
    const AT* __restrict__ A, const __bf16* __restrict__ W,
    const __bf16* __restrict__ bias, __bf16* __restrict__ C,
    int M, int N, int K) {
  constexpr int BM = 128, BN = 128, BK = 32;
  __shared__ __bf16 ldsA[2][BM * BK];      // 2 x 8 KB
  __shared__ __bf16 ldsBt[2][BN * BK];     // 2 x 8 KB, [col][k]

  const int tid  = threadIdx.x;
  const int lane = tid & 31;
  const int wave = tid >> 5;
  const int hi   = lane >> 4;              // lane half
  const int ln   = lane & 15;
  const int m0   = blockIdx.y * BM;
  const int n0   = blockIdx.x * BN;
  const int wm   = (wave >> 1) * 32;
  const int wn   = (wave & 1) * 64;

  // staging coords
  const int arow = tid >> 1, acb = (tid & 1) * 16;   // A: 128 rows x 2 chunks
  const int bkr  = (tid >> 5) * 4;                   // W: 4 k-rows per thread
  const int bcb  = (tid & 31) * 4;                   // 4 cols (lane-contig)

  v8f acc[2][4];
#pragma unroll
  for (int mi = 0; mi < 2; ++mi)
#pragma unroll
    for (int ni = 0; ni < 4; ++ni) acc[mi][ni] = vzero8();

  auto stage = [&](int k0, int buf) {
    // A tile (f32 converted to bf16 on the fly, or bf16 copy)
    stage16(A + (size_t)(m0 + arow) * K + k0 + acb, &ldsA[buf][arow * BK + acb]);
    // W tile transposed, packed: 4 rows x 4 cols -> 4 x ds_store_b64
    const __bf16* ws = W + (size_t)(k0 + bkr) * N + n0 + bcb;
    U4 r0, r1, r2, r3;
    r0.v = *(const v4bf*)ws;
    r1.v = *(const v4bf*)(ws + N);
    r2.v = *(const v4bf*)(ws + 2 * (size_t)N);
    r3.v = *(const v4bf*)(ws + 3 * (size_t)N);
#pragma unroll
    for (int j = 0; j < 4; ++j) {
      uint2 pv;
      pv.x = (u32)r0.s[j] | ((u32)r1.s[j] << 16);
      pv.y = (u32)r2.s[j] | ((u32)r3.s[j] << 16);
      *(uint2*)&ldsBt[buf][(bcb + j) * BK + bkr] = pv;
    }
  };

  stage(0, 0);
  __syncthreads();

  int buf = 0;
  for (int k0 = 0; k0 < K; k0 += BK, buf ^= 1) {
    // fragments for current tile FIRST (WMMA won't wait on next-tile staging)
    v16bf afrag[2], bfrag[4];
#pragma unroll
    for (int mi = 0; mi < 2; ++mi) {
      const int row = wm + mi * 16 + ln;
      F16Frag f;
      f.h[0] = *(const v8bf*)&ldsA[buf][row * BK + hi * 16];
      f.h[1] = *(const v8bf*)&ldsA[buf][row * BK + hi * 16 + 8];
      afrag[mi] = f.v;
    }
#pragma unroll
    for (int ni = 0; ni < 4; ++ni) {
      const int col = wn + ni * 16 + ln;
      bfrag[ni] = *(const v16bf*)&ldsBt[buf][col * BK + hi * 16];
    }
    // stage next tile into the other buffer (overlaps with WMMA burst)
    if (k0 + BK < K) stage(k0 + BK, buf ^ 1);

#pragma unroll
    for (int mi = 0; mi < 2; ++mi)
#pragma unroll
      for (int ni = 0; ni < 4; ++ni)
        acc[mi][ni] = wmma_bf16(afrag[mi], bfrag[ni], acc[mi][ni]);

    __syncthreads();
  }

  // epilogue: C layout: lane ln = column, VGPR r = row (r + 8*hi)
#pragma unroll
  for (int mi = 0; mi < 2; ++mi)
#pragma unroll
    for (int ni = 0; ni < 4; ++ni) {
      const int col = n0 + wn + ni * 16 + ln;
      float bb = 0.0f;
      if constexpr (HAS_BIAS) bb = (float)bias[col];
#pragma unroll
      for (int r = 0; r < 8; ++r) {
        const int row = m0 + wm + mi * 16 + r + 8 * hi;
        C[(size_t)row * N + col] = (__bf16)(acc[mi][ni][r] + bb);
      }
    }
}

// ---------------------------------------------------------------------------
// In-place RoPE on q [B,S,NH,HD] and cache_k [B,S,NKV,HD]
// ---------------------------------------------------------------------------
__global__ __launch_bounds__(256) void rope_kernel(__bf16* __restrict__ qb,
                                                   __bf16* __restrict__ kb,
                                                   const int* __restrict__ pos) {
  constexpr int B = 2, S = 1024, NH = 32, NKV = 8, HD = 128;
  const int NQ = B * S * NH * (HD / 2);
  const int NK = B * S * NKV * (HD / 2);
  int idx = blockIdx.x * blockDim.x + threadIdx.x;
  if (idx >= NQ + NK) return;
  __bf16* buf;
  int heads, li;
  if (idx < NQ) { buf = qb; heads = NH; li = idx; }
  else          { buf = kb; heads = NKV; li = idx - NQ; }
  const int i  = li & 63;
  const int hh = (li >> 6) % heads;
  const int s  = (li / (64 * heads)) % S;
  const int b  = li / (64 * heads * S);
  const float p = (float)pos[b * S + s];
  // inv_freq = theta^(-2i/HD) = exp(-i * ln(1e6)/64)
  const float invf = __expf(-(float)i * (13.8155106f / 64.0f));
  const float fr = p * invf;
  float sn, cs;
  __sincosf(fr, &sn, &cs);
  const size_t base = ((size_t)((b * S + s) * heads) + hh) * HD;
  const float x1 = (float)buf[base + i];
  const float x2 = (float)buf[base + i + 64];
  buf[base + i]      = (__bf16)(x1 * cs - x2 * sn);
  buf[base + i + 64] = (__bf16)(x2 * cs + x1 * sn);
}

// ---------------------------------------------------------------------------
// Flash attention (causal, GQA 4:1). Grid (S/128, NH, B), block 256 (8 waves).
// Wave owns 16 query rows; K/V tiles of 64 keys staged in LDS.
// K tile: async global->LDS copy when available. V tile: packed transpose.
// ---------------------------------------------------------------------------
__global__ __launch_bounds__(256) void attn_kernel(
    const __bf16* __restrict__ q, const __bf16* __restrict__ kc,
    const __bf16* __restrict__ vc, __bf16* __restrict__ o) {
  constexpr int S = 1024, NH = 32, NKV = 8, HD = 128, BK = 64;
  __shared__ __bf16 ldsK[BK * HD];        // [key][hd]  16 KB
  __shared__ __bf16 ldsVt[HD * BK];       // [hd][key]  16 KB
  __shared__ __bf16 ldsP[8 * 16 * 64];    // per-wave P 16 KB

  const int tid  = threadIdx.x;
  const int lane = tid & 31;
  const int wave = tid >> 5;
  const int hi   = lane >> 4;
  const int ln   = lane & 15;
  const int h    = blockIdx.y;
  const int b    = blockIdx.z;
  const int hkv  = h >> 2;                      // NH/NKV = 4
  const int q0   = blockIdx.x * 128 + wave * 16;
  const float scale = 0.08838834764831845f;     // 1/sqrt(128)

  // ---- Q A-fragments (held for the whole K loop) ----
  v16bf qf[4];
  {
    const size_t base = ((size_t)(b * S + q0 + ln) * NH + h) * HD;
#pragma unroll
    for (int kk = 0; kk < 4; ++kk) {
      F16Frag f;
      f.h[0] = *(const v8bf*)&q[base + kk * 32 + hi * 16];
      f.h[1] = *(const v8bf*)&q[base + kk * 32 + hi * 16 + 8];
      qf[kk] = f.v;
    }
  }

  v8f oacc[8];
#pragma unroll
  for (int nt = 0; nt < 8; ++nt) oacc[nt] = vzero8();
  float mrow[8], lrow[8];
#pragma unroll
  for (int r = 0; r < 8; ++r) { mrow[r] = -3.0e38f; lrow[r] = 0.0f; }

  const int ktmax = blockIdx.x * 2 + 1;   // causal: keys <= block's last row
  // staging coords
  const int kkey = tid >> 2, khdb = (tid & 3) * 32;   // K: 32 hd per thread
  const int vhd  = (tid & 31) * 4;                    // V: 4 hd (lane-contig)
  const int vkey = (tid >> 5) * 8;                    //    8 keys per thread

  for (int kt = 0; kt <= ktmax; ++kt) {
    const int k0 = kt * BK;
    // ---- stage K (row-major copy; async global->LDS path when supported) ----
    {
      const size_t gk = ((size_t)(b * S + k0 + kkey) * NKV + hkv) * HD + khdb;
#if USE_ASYNC_LDS
      as1_v4i* g = (as1_v4i*)(const void*)&kc[gk];
      as3_v4i* l = (as3_v4i*)(void*)&ldsK[kkey * HD + khdb];
#pragma unroll
      for (int j = 0; j < 4; ++j)
        __builtin_amdgcn_global_load_async_to_lds_b128(g + j, l + j, 0, 0);
#else
      v8bf a0 = *(const v8bf*)&kc[gk],      a1 = *(const v8bf*)&kc[gk + 8];
      v8bf a2 = *(const v8bf*)&kc[gk + 16], a3 = *(const v8bf*)&kc[gk + 24];
      *(v8bf*)&ldsK[kkey * HD + khdb]      = a0;
      *(v8bf*)&ldsK[kkey * HD + khdb + 8]  = a1;
      *(v8bf*)&ldsK[kkey * HD + khdb + 16] = a2;
      *(v8bf*)&ldsK[kkey * HD + khdb + 24] = a3;
#endif
    }
    // ---- stage V transposed [hd][key], packed -> 4 x ds_store_b128 ----
    {
      U4 rv[8];
#pragma unroll
      for (int i = 0; i < 8; ++i)
        rv[i].v = *(const v4bf*)&vc[((size_t)(b * S + k0 + vkey + i) * NKV + hkv) * HD + vhd];
#pragma unroll
      for (int j = 0; j < 4; ++j) {
        uint4 pv;
        pv.x = (u32)rv[0].s[j] | ((u32)rv[1].s[j] << 16);
        pv.y = (u32)rv[2].s[j] | ((u32)rv[3].s[j] << 16);
        pv.z = (u32)rv[4].s[j] | ((u32)rv[5].s[j] << 16);
        pv.w = (u32)rv[6].s[j] | ((u32)rv[7].s[j] << 16);
        *(uint4*)&ldsVt[(vhd + j) * BK + vkey] = pv;
      }
    }
#if USE_ASYNC_LDS
    __builtin_amdgcn_s_wait_asynccnt(0);
#endif
    __syncthreads();

    // ---- S = Q K^T : 4 key sub-tiles x 4 hd k-steps ----
    v8f sacc[4];
#pragma unroll
    for (int ni = 0; ni < 4; ++ni) {
      sacc[ni] = vzero8();
#pragma unroll
      for (int kk = 0; kk < 4; ++kk) {
        v16bf bf = *(const v16bf*)&ldsK[(ni * 16 + ln) * HD + kk * 32 + hi * 16];
        sacc[ni] = wmma_bf16(qf[kk], bf, sacc[ni]);
      }
    }

    // ---- online softmax (row = r + 8*hi lives on a 16-lane half) ----
    float alpha[8];
#pragma unroll
    for (int r = 0; r < 8; ++r) {
      const int srow = q0 + r + 8 * hi;
      float tmax = -3.0e38f;
#pragma unroll
      for (int ni = 0; ni < 4; ++ni) {
        const int kcol = k0 + ni * 16 + ln;
        float sv = sacc[ni][r] * scale + ((srow >= kcol) ? 0.0f : -1.0e9f);
        sacc[ni][r] = sv;
        tmax = fmaxf(tmax, sv);
      }
#pragma unroll
      for (int ms = 1; ms <= 8; ms <<= 1)
        tmax = fmaxf(tmax, __shfl_xor(tmax, ms, 32));
      const float mnew = fmaxf(mrow[r], tmax);
      const float a = __expf(mrow[r] - mnew);
      mrow[r] = mnew;
      alpha[r] = a;
      float rsum = 0.0f;
#pragma unroll
      for (int ni = 0; ni < 4; ++ni) {
        float pv = __expf(sacc[ni][r] - mnew);
        sacc[ni][r] = pv;
        rsum += pv;
      }
#pragma unroll
      for (int ms = 1; ms <= 8; ms <<= 1) rsum += __shfl_xor(rsum, ms, 32);
      lrow[r] = lrow[r] * a + rsum;
    }
#pragma unroll
    for (int nt = 0; nt < 8; ++nt)
#pragma unroll
      for (int r = 0; r < 8; ++r) oacc[nt][r] *= alpha[r];

    // ---- restage P (C-layout -> A-fragment layout) in per-wave LDS ----
    __bf16* pw = &ldsP[wave * 16 * 64];
#pragma unroll
    for (int ni = 0; ni < 4; ++ni)
#pragma unroll
      for (int r = 0; r < 8; ++r)
        pw[(r + 8 * hi) * 64 + ni * 16 + ln] = (__bf16)sacc[ni][r];

    // ---- O += P V : 2 key k-steps x 8 hd sub-tiles ----
#pragma unroll
    for (int kk2 = 0; kk2 < 2; ++kk2) {
      F16Frag pf;
      pf.h[0] = *(const v8bf*)&pw[ln * 64 + kk2 * 32 + hi * 16];
      pf.h[1] = *(const v8bf*)&pw[ln * 64 + kk2 * 32 + hi * 16 + 8];
#pragma unroll
      for (int nt = 0; nt < 8; ++nt) {
        v16bf vf = *(const v16bf*)&ldsVt[(nt * 16 + ln) * BK + kk2 * 32 + hi * 16];
        oacc[nt] = wmma_bf16(pf.v, vf, oacc[nt]);
      }
    }
    __syncthreads();
  }

  // ---- normalize + store [B,S,NH,HD] ----
#pragma unroll
  for (int r = 0; r < 8; ++r) {
    const float invl = 1.0f / lrow[r];
    const size_t base = ((size_t)(b * S + q0 + r + 8 * hi) * NH + h) * HD;
#pragma unroll
    for (int nt = 0; nt < 8; ++nt)
      o[base + nt * 16 + ln] = (__bf16)(oacc[nt][r] * invl);
  }
}

// ---------------------------------------------------------------------------
extern "C" void kernel_launch(void* const* d_in, const int* in_sizes, int n_in,
                              void* d_out, int out_size, void* d_ws, size_t ws_size,
                              hipStream_t stream) {
  (void)in_sizes; (void)n_in; (void)out_size; (void)ws_size;
  constexpr int B = 2, S = 1024, H = 4096, NKV = 8, HD = 128;
  constexpr int M = B * S;                  // 2048
  constexpr int KVD = NKV * HD;             // 1024

  const float*  hidden = (const float*)d_in[0];
  // d_in[1] = attention_mask (causal; applied analytically in attn_kernel)
  const int*    pos    = (const int*)d_in[2];
  const __bf16* Wq = (const __bf16*)d_in[3];
  const __bf16* bq = (const __bf16*)d_in[4];
  const __bf16* Wk = (const __bf16*)d_in[5];
  const __bf16* bk = (const __bf16*)d_in[6];
  const __bf16* Wv = (const __bf16*)d_in[7];
  const __bf16* bv = (const __bf16*)d_in[8];
  const __bf16* Wo = (const __bf16*)d_in[9];

  __bf16* out     = (__bf16*)d_out;                       // [B,S,H]
  __bf16* cache_k = out + (size_t)M * H;                  // [B,S,NKV,HD]
  __bf16* cache_v = cache_k + (size_t)M * KVD;            // [B,S,NKV,HD]

  __bf16* qbuf    = (__bf16*)d_ws;                        // [B,S,H]
  __bf16* attnbuf = qbuf + (size_t)M * H;                 // [B,S,H]

  dim3 blk(256);
  // Q/K/V projections (f32 activations -> bf16 staged, bf16 WMMA, +bias)
  gemm_wmma<float, true><<<dim3(H / 128, M / 128), blk, 0, stream>>>(
      hidden, Wq, bq, qbuf, M, H, H);
  gemm_wmma<float, true><<<dim3(KVD / 128, M / 128), blk, 0, stream>>>(
      hidden, Wk, bk, cache_k, M, KVD, H);
  gemm_wmma<float, true><<<dim3(KVD / 128, M / 128), blk, 0, stream>>>(
      hidden, Wv, bv, cache_v, M, KVD, H);
  // RoPE in-place on q and cache_k
  const int npairs = B * S * (32 + NKV) * (HD / 2);
  rope_kernel<<<(npairs + 255) / 256, blk, 0, stream>>>(qbuf, cache_k, pos);
  // flash attention -> attnbuf
  attn_kernel<<<dim3(S / 128, 32, B), blk, 0, stream>>>(qbuf, cache_k, cache_v,
                                                        attnbuf);
  // O projection (bf16 A, no bias)
  gemm_wmma<__bf16, false><<<dim3(H / 128, M / 128), blk, 0, stream>>>(
      attnbuf, Wo, (const __bf16*)nullptr, out, M, H, H);
}
